// SupConLoss_80650895884987
// MI455X (gfx1250) — matrix operand order repaired and checked by reading, not compiled, not run on hardware
//
#include <hip/hip_runtime.h>
#include <math.h>

// ---------------------------------------------------------------------------
// SupConLoss fused kernels for gfx1250 (MI455X, wave32, WMMA, async->LDS)
//   feat1 [2048,512] f32, feat2 [2048,512] f32, labels [2048] i64 -> scalar f32
// ---------------------------------------------------------------------------

typedef _Float16 half16 __attribute__((ext_vector_type(16)));
typedef _Float16 half8  __attribute__((ext_vector_type(8)));
typedef float    float8 __attribute__((ext_vector_type(8)));

#define TAU      0.07f
#define EPSV     1e-8f
#define B_ROWS   2048
#define D_DIM    512
#define N_ROWS   4096            // 2*B
#define M_BLOCK  128             // rows per block in simloss (8 waves x 16 rows)
#define JSPLIT   8               // column splits across grid.y
#define JCOLS    (N_ROWS / JSPLIT)   // 512 columns per block
#define JTILES   (JCOLS / 16)        // 32 tiles of 16 columns
#define BTILE_BYTES (16 * D_DIM * 2) // 16 KB per B tile

// ---------------------------------------------------------------------------
// Kernel 1: L2-normalize each row of concat(feat1,feat2), emit f16 matrix zh.
// ---------------------------------------------------------------------------
__global__ void __launch_bounds__(256)
supcon_normalize_kernel(const float* __restrict__ f1, const float* __restrict__ f2,
                        _Float16* __restrict__ zh) {
    const int r = blockIdx.x;
    const float* src = (r < B_ROWS) ? (f1 + (size_t)r * D_DIM)
                                    : (f2 + (size_t)(r - B_ROWS) * D_DIM);
    const int t = threadIdx.x;
    const float x0 = src[t];
    const float x1 = src[t + 256];
    float ss = x0 * x0 + x1 * x1;

    #pragma unroll
    for (int off = 16; off > 0; off >>= 1) ss += __shfl_xor(ss, off, 32);

    __shared__ float red[8];
    const int w = t >> 5, l = t & 31;
    if (l == 0) red[w] = ss;
    __syncthreads();
    const float tot = red[0] + red[1] + red[2] + red[3] +
                      red[4] + red[5] + red[6] + red[7];
    const float rn = rsqrtf(tot);

    zh[(size_t)r * D_DIM + t]       = (_Float16)(x0 * rn);
    zh[(size_t)r * D_DIM + t + 256] = (_Float16)(x1 * rn);
}

// ---------------------------------------------------------------------------
// Kernel 2: fused  sim = (zh zh^T)/tau  ->  exp  ->  masked row sums.
// grid = (N_ROWS/M_BLOCK, JSPLIT); block = 256 threads = 8 waves.
//
// Pipeline per block:
//   - A fragments (16 rows x 512 K per wave) preloaded into 128 VGPRs/lane
//   - B tile (16 cols x 512 K, 16 KB) double-buffered in LDS via
//     global_load_async_to_lds_b128 (ASYNCcnt), overlapped with the 16-WMMA
//     K chain of the previous tile; s_wait_asynccnt 0 + barrier per tile
//   - B fragment register double-buffer hides ds_load latency per WMMA
//
// CDNA5 wave32 fragment layouts as used here (k contiguous per lane half, not ISA table order):
//   A (16x32 f16): lane l: row m = l&15, group g = l>>4;
//                  element e: k = g*16 + e
//   B (32x16 f16): lane l: col n = l&15, group g = l>>4; element e: k = g*16+e
//   C/D (16x16 f32): lane l: col n = l&15; element v: row m = (l>>4)*8 + v
// ---------------------------------------------------------------------------
__global__ void __launch_bounds__(256)
supcon_simloss_kernel(const _Float16* __restrict__ zh,
                      const long long* __restrict__ labels,
                      float* __restrict__ num_part,
                      float* __restrict__ den_part) {
    __shared__ _Float16 ldsB[2][16 * D_DIM];   // 2 x 16 KB double buffer

    const int tid = threadIdx.x;
    const int w = tid >> 5;         // wave in block (0..7)
    const int l = tid & 31;         // lane in wave
    const int g = l >> 4;           // half-wave group
    const int n = l & 15;           // output column in tile / A row in strip
    const int bx = blockIdx.x;      // row strip
    const int by = blockIdx.y;      // column split

    const int rbase = bx * M_BLOCK + w * 16 + g * 8;   // first of lane's 8 rows

    int   rlab[8];
    float num_acc[8], den_acc[8];
    #pragma unroll
    for (int v = 0; v < 8; ++v) {
        rlab[v] = (int)labels[(rbase + v) & (B_ROWS - 1)];
        num_acc[v] = 0.0f;
        den_acc[v] = 0.0f;
    }

    // --- preload this lane's 16 A fragments (K = 512) into registers ---
    const _Float16* __restrict__ arow =
        zh + (size_t)(bx * M_BLOCK + w * 16 + n) * D_DIM;
    half16 afrag[16];
    #pragma unroll
    for (int kt = 0; kt < 16; ++kt) {
        const int kb = kt * 32;
        const half8 alo = *(const half8*)(arow + kb + g * 16);
        const half8 ahi = *(const half8*)(arow + kb + g * 16 + 8);
        #pragma unroll
        for (int e = 0; e < 8; ++e) { afrag[kt][e] = alo[e]; afrag[kt][e + 8] = ahi[e]; }
    }

    // --- async B-tile staging: global -> LDS, ASYNCcnt-tracked ---
    const uint32_t ldsBase = (uint32_t)(uintptr_t)(&ldsB[0][0]);
    const uint64_t zhBase  = (uint64_t)(uintptr_t)zh;
    auto stage_async = [&](int jt, int buf) {
        #pragma unroll
        for (int i = 0; i < 4; ++i) {
            const int c = tid * 4 + i;                 // 1024 16B chunks / tile
            const uint32_t ldsaddr = ldsBase + (uint32_t)buf * BTILE_BYTES
                                             + (uint32_t)c * 16u;
            const uint64_t gaddr = zhBase +
                (uint64_t)((uint32_t)(by * JCOLS + jt * 16 + (c >> 6)) * (D_DIM * 2u)
                           + (uint32_t)(c & 63) * 16u);
            asm volatile("global_load_async_to_lds_b128 %0, %1, off"
                         :: "v"(ldsaddr), "v"(gaddr)
                         : "memory");
        }
    };

    stage_async(0, 0);
    asm volatile("s_wait_asynccnt 0x0" ::: "memory");
    __syncthreads();

    for (int jt = 0; jt < JTILES; ++jt) {
        // kick off next tile's async copy into the other buffer
        if (jt + 1 < JTILES) stage_async(jt + 1, (jt + 1) & 1);

        // --- K accumulation: 16 chained v_wmma_f32_16x16x32_f16 ---
        const _Float16* bb = &ldsB[jt & 1][n * D_DIM + g * 16];
        half16 bcur = *(const half16*)(bb);
        float8 acc = {};
        #pragma unroll
        for (int kt = 0; kt < 16; ++kt) {
            const half16 bk = bcur;
            if (kt < 15) bcur = *(const half16*)(bb + (kt + 1) * 32);
            acc = __builtin_amdgcn_wmma_f32_16x16x32_f16(
                      false, afrag[kt], false, bk, (short)0, acc, false, false);
        }

        // --- exp + masks + accumulate ---
        const int jbase = by * JCOLS + jt * 16;
        const int col   = jbase + n;
        const int clab  = (int)labels[col & (B_ROWS - 1)];
        #pragma unroll
        for (int v = 0; v < 8; ++v) {
            const int   row = rbase + v;
            const float e   = __expf(acc[v] * (1.0f / TAU));
            den_acc[v] += e;                                   // denom keeps self
            if ((clab == rlab[v]) && (row != col)) num_acc[v] += e;
        }

        // next tile's async writes must land before anyone reads them
        asm volatile("s_wait_asynccnt 0x0" ::: "memory");
        __syncthreads();
    }

    // --- reduce across the 16 lanes sharing the same rows ---
    #pragma unroll
    for (int v = 0; v < 8; ++v) {
        float nu = num_acc[v], de = den_acc[v];
        #pragma unroll
        for (int off = 1; off < 16; off <<= 1) {
            nu += __shfl_xor(nu, off, 32);
            de += __shfl_xor(de, off, 32);
        }
        num_acc[v] = nu;
        den_acc[v] = de;
    }
    if (n == 0) {
        #pragma unroll
        for (int v = 0; v < 8; ++v) {
            const int row = rbase + v;
            num_part[(size_t)by * N_ROWS + row] = num_acc[v];
            den_part[(size_t)by * N_ROWS + row] = den_acc[v];
        }
    }
}

// ---------------------------------------------------------------------------
// Kernel 3: per-row loss + mean. Single block, deterministic tree reduce.
// ---------------------------------------------------------------------------
__global__ void __launch_bounds__(256)
supcon_reduce_kernel(const float* __restrict__ num_part,
                     const float* __restrict__ den_part,
                     float* __restrict__ out) {
    const int t = threadIdx.x;
    float lsum = 0.0f;
    for (int r = t; r < N_ROWS; r += 256) {
        float nu = 0.0f, de = 0.0f;
        #pragma unroll
        for (int p = 0; p < JSPLIT; ++p) {
            nu += num_part[(size_t)p * N_ROWS + r];
            de += den_part[(size_t)p * N_ROWS + r];
        }
        lsum += -__logf(nu / (de + EPSV) + EPSV);
    }
    __shared__ float red[256];
    red[t] = lsum;
    __syncthreads();
    for (int s = 128; s > 0; s >>= 1) {
        if (t < s) red[t] += red[t + s];
        __syncthreads();
    }
    if (t == 0) out[0] = red[0] / (float)N_ROWS;
}

// ---------------------------------------------------------------------------
// Launch
// ---------------------------------------------------------------------------
extern "C" void kernel_launch(void* const* d_in, const int* in_sizes, int n_in,
                              void* d_out, int out_size, void* d_ws, size_t ws_size,
                              hipStream_t stream) {
    (void)in_sizes; (void)n_in; (void)out_size; (void)ws_size;
    const float*     feat1  = (const float*)d_in[0];
    const float*     feat2  = (const float*)d_in[1];
    const long long* labels = (const long long*)d_in[2];

    char* ws = (char*)d_ws;
    _Float16* zh = (_Float16*)ws;                               // 4 MB
    const size_t zh_bytes = (size_t)N_ROWS * D_DIM * sizeof(_Float16);
    float* num_part = (float*)(ws + zh_bytes);                  // 128 KB
    float* den_part = num_part + (size_t)JSPLIT * N_ROWS;       // 128 KB

    supcon_normalize_kernel<<<N_ROWS, 256, 0, stream>>>(feat1, feat2, zh);

    dim3 grid2(N_ROWS / M_BLOCK, JSPLIT);
    supcon_simloss_kernel<<<grid2, 256, 0, stream>>>(zh, labels, num_part, den_part);

    supcon_reduce_kernel<<<1, 256, 0, stream>>>(num_part, den_part, (float*)d_out);
}
